// Seq2Seq_65163243815385
// MI455X (gfx1250) — matrix operand order repaired and checked by reading, not compiled, not run on hardware
//
#include <hip/hip_runtime.h>
#include <hip/hip_bf16.h>
#include <stdint.h>

// ---------------------------------------------------------------------------
// Seq2Seq GRU encoder/decoder + vocab projection for MI455X (gfx1250, wave32)
// GEMMs: v_wmma_f32_16x16x32_f16; weight stream via TDM tensor_load_to_lds
// with per-wave double buffering (s_wait_tensorcnt), copy-free 2x-unrolled
// register ping-pong for A fragments.
// ---------------------------------------------------------------------------

typedef __attribute__((ext_vector_type(16))) _Float16 v16h;
typedef __attribute__((ext_vector_type(8)))  _Float16 v8h;
typedef __attribute__((ext_vector_type(8)))  float    v8f;
typedef __attribute__((ext_vector_type(4)))  unsigned int v4u;
typedef __attribute__((ext_vector_type(4)))  int      v4i;
typedef __attribute__((ext_vector_type(8)))  int      v8i_t;

#define SB 64        // batch
#define ST 64        // time
#define SV 32000     // vocab
#define SE 512       // emb dim
#define SH 1024      // hidden dim (enc == dec)
#define SG 3072      // 3*H gate dim

#if defined(__has_builtin)
#  if __has_builtin(__builtin_amdgcn_tensor_load_to_lds)
#    define HAVE_TDM 1
#  else
#    define HAVE_TDM 0
#  endif
#else
#  define HAVE_TDM 0
#endif

#if __has_include(<hip/amd_detail/amd_gfx1250_TDM.h>)
#  define TDM_ARITY6 1
#else
#  define TDM_ARITY6 0
#endif

// ---------------------------------------------------------------------------
// Fragment helpers (contiguous k order, same for A and B, wave32)
// A 16x32 f16 (row-major, lda=K):
//   lanes 0-15 : M=lane,    halves K {0..15}
//   lanes 16-31: M=lane-16, halves K {16..31}
// B 32x16 f16 from row-major [N,K] weights:
//   lanes 0-15 : N=lane,    halves K {0..15};  lanes 16-31: K {16..31}
// C/D v8f: row = i + 8*(lane>=16), col = lane&15
// ---------------------------------------------------------------------------
__device__ __forceinline__ v16h load_a_frag(const _Float16* __restrict__ p) {
    v16h a;
    ((v8h*)&a)[0] = *(const v8h*)(p);
    ((v8h*)&a)[1] = *(const v8h*)(p + 8);
    return a;
}

__device__ __forceinline__ v8f wmma_f16(v16h a, v16h b, v8f c) {
    return __builtin_amdgcn_wmma_f32_16x16x32_f16(
        false, a, false, b, (short)0, c, false, false);
}

// ---------------------------------------------------------------------------
// TDM: DMA a 16-row x 32-half f16 strip of a row-major [N,K] matrix into LDS.
// D# per ISA §8.3/8.4: count=1 | lds_addr | global_addr | type=2;
// data_size=2B; tensor 32x16 anchored at tile base; tensor_dim0_stride = K.
// LDS destination: 16 rows x 64 B densely packed (1 KB).
// ---------------------------------------------------------------------------
__device__ __forceinline__ void tdm_load_16x32(_Float16* lds_ptr,
                                               const _Float16* gsrc,
                                               int K, int lane) {
#if HAVE_TDM
    const unsigned long long ga = (unsigned long long)(uintptr_t)gsrc;
    const unsigned lds_off = (unsigned)(uintptr_t)lds_ptr;  // low 32b = LDS offset
    v4u g0;
    g0.x = 1u;                                     // count=1 (valid user D#)
    g0.y = lds_off;                                // lds_addr (bytes)
    g0.z = (unsigned)(ga & 0xFFFFFFFFu);           // global_addr[31:0]
    g0.w = (unsigned)((ga >> 32) & 0x01FFFFFFu)    // global_addr[56:32]
         | 0x80000000u;                            // type=2 ("image")
    v8i_t g1;
    g1[0] = (int)(1u << 16);                       // data_size=2B, wg_mask=0
    g1[1] = (int)(32u << 16);                      // tensor_dim0 = 32
    g1[2] = (int)(16u << 16);                      // tensor_dim1 = 16
    g1[3] = (int)(32u << 16);                      // tile_dim0 = 32
    g1[4] = 16;                                    // tile_dim1 = 16
    g1[5] = K;                                     // tensor_dim0_stride = K
    g1[6] = 0;
    g1[7] = 0;
    v4i g2 = {0, 0, 0, 0};                         // 2D: groups 2/3 unused
    v4i g3 = {0, 0, 0, 0};
#if TDM_ARITY6
    v8i_t gz = {0, 0, 0, 0, 0, 0, 0, 0};
    __builtin_amdgcn_tensor_load_to_lds(g0, g1, g2, g3, gz, 0);
#else
    __builtin_amdgcn_tensor_load_to_lds(g0, g1, g2, g3, 0);
#endif
#else
    // Cooperative fallback: 32 lanes x 32 B = 1 KB
    const int r = lane >> 1;
    const int p = (lane & 1) << 4;
    *(v16h*)(lds_ptr + r * 32 + p) = *(const v16h*)(gsrc + (size_t)r * K + p);
#endif
}

__device__ __forceinline__ void wait_tensor_le1() {
#if HAVE_TDM
    __builtin_amdgcn_s_wait_tensorcnt(1);
#endif
    __asm__ volatile("" ::: "memory");
}
__device__ __forceinline__ void wait_tensor_0() {
#if HAVE_TDM
    __builtin_amdgcn_s_wait_tensorcnt(0);
#endif
    __asm__ volatile("" ::: "memory");
}

// ---------------------------------------------------------------------------
// Generic TN GEMM: C[M,N] = A[M,K](f16,rm) * B[N,K](f16,rm)^T + bias[N]
// 256 threads = 8 waves; WG tile 64(M) x 128(N); wave w owns 16-col strip.
// Steady state is 2x unrolled: stage s in {buf0,aA}, stage s+1 in {buf1,aB};
// each body issues the +1 stage then consumes its own under its own names,
// so there are no register-rotation moves. Requires K/32 even (holds for
// K in {512, 1024, 1536}).  grid = (N/128, M/64)
// ---------------------------------------------------------------------------
__global__ __launch_bounds__(256)
void gemm_tn_wmma(const _Float16* __restrict__ A,
                  const _Float16* __restrict__ B,
                  const float* __restrict__ bias,
                  float* __restrict__ C,
                  int M, int N, int K) {
    __shared__ _Float16 bstage[8][2][16 * 32];   // 16 KB: per-wave double buffer

    const int lane = threadIdx.x & 31;
    const int wave = threadIdx.x >> 5;
    const int n0   = blockIdx.x * 128 + wave * 16;
    const int m0   = blockIdx.y * 64;

    const int arow = lane & 15;
    const int asel = (lane >> 4) << 4;    // 0 or 16
    const int bsel = (lane >> 4) << 4;    // 0 or 16
    const int boff = (lane & 15) * 32 + bsel;

    const _Float16* __restrict__ Ap = A + (size_t)(m0 + arow) * K + asel;
    const _Float16* __restrict__ Bt = B + (size_t)n0 * K;   // wave's 16-row strip

    v8f acc[4] = {v8f{}, v8f{}, v8f{}, v8f{}};
    v16h aA[4], aB[4];

    const int nsteps = K >> 5;   // even

    // prologue: stage 0 -> {buf0, aA}
    tdm_load_16x32(&bstage[wave][0][0], Bt, K, lane);
#pragma unroll
    for (int mt = 0; mt < 4; ++mt)
        aA[mt] = load_a_frag(Ap + (size_t)(mt * 16) * K);

    for (int s = 0; s + 2 <= nsteps - 1; s += 2) {
        const int k1 = (s + 1) << 5;
        const int k2 = (s + 2) << 5;
        // body A: issue stage s+1 -> {buf1, aB}; consume stage s
        tdm_load_16x32(&bstage[wave][1][0], Bt + k1, K, lane);
        __builtin_prefetch((const void*)(Ap + k2 + 256), 0, 1);
#pragma unroll
        for (int mt = 0; mt < 4; ++mt)
            aB[mt] = load_a_frag(Ap + (size_t)(mt * 16) * K + k1);
        wait_tensor_le1();
        {
            v16h bf = *(const v16h*)(&bstage[wave][0][boff]);
#pragma unroll
            for (int mt = 0; mt < 4; ++mt)
                acc[mt] = wmma_f16(aA[mt], bf, acc[mt]);
        }
        // body B: issue stage s+2 -> {buf0, aA}; consume stage s+1
        tdm_load_16x32(&bstage[wave][0][0], Bt + k2, K, lane);
#pragma unroll
        for (int mt = 0; mt < 4; ++mt)
            aA[mt] = load_a_frag(Ap + (size_t)(mt * 16) * K + k2);
        wait_tensor_le1();
        {
            v16h bf = *(const v16h*)(&bstage[wave][1][boff]);
#pragma unroll
            for (int mt = 0; mt < 4; ++mt)
                acc[mt] = wmma_f16(aB[mt], bf, acc[mt]);
        }
    }
    // epilogue: stage nsteps-2 is in {buf0, aA}; stage nsteps-1 still to issue
    {
        const int kl = (nsteps - 1) << 5;
        tdm_load_16x32(&bstage[wave][1][0], Bt + kl, K, lane);
#pragma unroll
        for (int mt = 0; mt < 4; ++mt)
            aB[mt] = load_a_frag(Ap + (size_t)(mt * 16) * K + kl);
        wait_tensor_le1();
        {
            v16h bf = *(const v16h*)(&bstage[wave][0][boff]);
#pragma unroll
            for (int mt = 0; mt < 4; ++mt)
                acc[mt] = wmma_f16(aA[mt], bf, acc[mt]);
        }
        wait_tensor_0();
        {
            v16h bf = *(const v16h*)(&bstage[wave][1][boff]);
#pragma unroll
            for (int mt = 0; mt < 4; ++mt)
                acc[mt] = wmma_f16(aB[mt], bf, acc[mt]);
        }
    }

    const int   ccol  = n0 + (lane & 15);
    const float bv    = bias ? bias[ccol] : 0.0f;
    const int   rbase = m0 + ((lane >> 4) << 3);
#pragma unroll
    for (int mt = 0; mt < 4; ++mt)
#pragma unroll
        for (int i = 0; i < 8; ++i)
            C[(size_t)(rbase + mt * 16 + i) * N + ccol] = acc[mt][i] + bv;
}

// ---------------------------------------------------------------------------
// Fused GRU timestep: gh = h16 @ Whh^T (WMMA -> LDS), then gating.
// K-loop is 2x unrolled with alternating register stages (SH/32 = 32, even).
// grid = 16 WGs x 64 gate-cols, block = 256.
// ---------------------------------------------------------------------------
__global__ __launch_bounds__(256)
void gru_step_wmma(const _Float16* __restrict__ h16_in,
                   const float*    __restrict__ h32_in,
                   const float*    __restrict__ gx,
                   int t,
                   const _Float16* __restrict__ Whh,   // [3H, H] f16
                   const float*    __restrict__ bhh,   // [3H]
                   _Float16* __restrict__ h16_out,
                   float*    __restrict__ h32_out,
                   _Float16* __restrict__ states16) {  // [B*T, H] or null
    __shared__ float ghs[3][64][64];          // 48 KB

    const int lane = threadIdx.x & 31;
    const int wave = threadIdx.x >> 5;
    const int j0   = blockIdx.x * 64;

    const int arow = lane & 15;
    const int asel = (lane >> 4) << 4;
    const int bsel = (lane >> 4) << 4;

    for (int tile = wave; tile < 48; tile += 8) {
        const int g  = tile >> 4;
        const int mt = (tile >> 2) & 3;
        const int nt = tile & 3;
        const int bcol = g * SH + j0 + (nt << 4) + (lane & 15);
        const _Float16* __restrict__ Bp = Whh + (size_t)bcol * SH + bsel;
        const _Float16* __restrict__ Ap =
            h16_in + (size_t)((mt << 4) + arow) * SH + asel;

        v8f acc = {};
        v16h a0 = load_a_frag(Ap);
        v16h b0 = *(const v16h*)(Bp);
        for (int k = 0; k + 64 <= SH - 32; k += 64) {
            v16h a1 = load_a_frag(Ap + k + 32);
            v16h b1 = *(const v16h*)(Bp + k + 32);
            acc = wmma_f16(a0, b0, acc);
            a0 = load_a_frag(Ap + k + 64);
            b0 = *(const v16h*)(Bp + k + 64);
            acc = wmma_f16(a1, b1, acc);
        }
        {
            v16h a1 = load_a_frag(Ap + SH - 32);
            v16h b1 = *(const v16h*)(Bp + SH - 32);
            acc = wmma_f16(a0, b0, acc);
            acc = wmma_f16(a1, b1, acc);
        }

        const int cb = (mt << 4) + ((lane >> 4) << 3);
        const int cc = (nt << 4) + (lane & 15);
#pragma unroll
        for (int i = 0; i < 8; ++i) ghs[g][cb + i][cc] = acc[i];
    }
    __syncthreads();

    for (int e = threadIdx.x; e < 64 * 64; e += 256) {
        const int b  = e >> 6;
        const int jj = e & 63;
        const int j  = j0 + jj;
        const float* __restrict__ gxr = gx + (size_t)(b * ST + t) * SG;
        const float xr = gxr[j];
        const float xz = gxr[SH + j];
        const float xn = gxr[2 * SH + j];
        const float hr = ghs[0][b][jj] + bhh[j];
        const float hz = ghs[1][b][jj] + bhh[SH + j];
        const float hn = ghs[2][b][jj] + bhh[2 * SH + j];
        const float r  = 1.0f / (1.0f + __expf(-(xr + hr)));
        const float z  = 1.0f / (1.0f + __expf(-(xz + hz)));
        const float nn = tanhf(xn + r * hn);
        const float hp = h32_in[b * SH + j];
        const float hv = (1.0f - z) * nn + z * hp;
        h32_out[b * SH + j] = hv;
        h16_out[b * SH + j] = (_Float16)hv;
        if (states16) states16[(size_t)(b * ST + t) * SH + j] = (_Float16)hv;
    }
}

// ---------------------------------------------------------------------------
// Small helper kernels
// ---------------------------------------------------------------------------
__global__ void cvt_f32_f16(const float* __restrict__ s,
                            _Float16* __restrict__ d, size_t n) {
    for (size_t i = blockIdx.x * (size_t)blockDim.x + threadIdx.x; i < n;
         i += (size_t)gridDim.x * blockDim.x)
        d[i] = (_Float16)s[i];
}

__global__ void build_a_enc(const int* __restrict__ x,
                            const float* __restrict__ emb,
                            _Float16* __restrict__ A) {
    const int r   = blockIdx.x;
    const int tok = x[r];
    const float* __restrict__ src = emb + (size_t)tok * SE;
    _Float16* __restrict__ dst = A + (size_t)r * SE;
    for (int i = threadIdx.x; i < SE; i += blockDim.x) dst[i] = (_Float16)src[i];
}

__global__ void build_a_dec(const int* __restrict__ labels,
                            const float* __restrict__ emb,
                            const _Float16* __restrict__ hfin,
                            _Float16* __restrict__ A) {
    const int r = blockIdx.x;
    const int b = r / ST, t = r % ST;
    const int tok = (t == 0) ? 1 : labels[b * ST + t - 1];
    const float* __restrict__ src = emb + (size_t)tok * SE;
    _Float16* __restrict__ dst = A + (size_t)r * (SE + SH);
    for (int i = threadIdx.x; i < SE; i += blockDim.x) dst[i] = (_Float16)src[i];
    const _Float16* __restrict__ ctx = hfin + (size_t)b * SH;
    for (int i = threadIdx.x; i < SH; i += blockDim.x) dst[SE + i] = ctx[i];
}

__global__ void init_h(float* __restrict__ h32, _Float16* __restrict__ h16,
                       const float* __restrict__ src) {
    const int i = blockIdx.x * blockDim.x + threadIdx.x;
    if (i < SB * SH) {
        const float v = src ? src[i & (SH - 1)] : 0.0f;
        h32[i] = v;
        h16[i] = (_Float16)v;
    }
}

// ---------------------------------------------------------------------------
// Host orchestration
// ---------------------------------------------------------------------------
extern "C" void kernel_launch(void* const* d_in, const int* in_sizes, int n_in,
                              void* d_out, int out_size, void* d_ws, size_t ws_size,
                              hipStream_t stream) {
    const int*   x        = (const int*)  d_in[0];
    const int*   labels   = (const int*)  d_in[1];
    const float* enc_emb  = (const float*)d_in[2];
    const float* enc_Wih  = (const float*)d_in[3];
    const float* enc_Whh  = (const float*)d_in[4];
    const float* enc_bih  = (const float*)d_in[5];
    const float* enc_bhh  = (const float*)d_in[6];
    const float* dec_emb  = (const float*)d_in[7];
    const float* dec_Wih  = (const float*)d_in[8];
    const float* dec_Whh  = (const float*)d_in[9];
    const float* dec_bih  = (const float*)d_in[10];
    const float* dec_bhh  = (const float*)d_in[11];
    const float* dec_init = (const float*)d_in[12];
    const float* lin_W    = (const float*)d_in[13];
    const float* lin_b    = (const float*)d_in[14];
    float* out = (float*)d_out;

    char* w = (char*)d_ws;
    auto carve = [&](size_t bytes) {
        char* p = w;
        w += (bytes + 255) & ~(size_t)255;
        return (void*)p;
    };
    _Float16* enc_Wih16 = (_Float16*)carve((size_t)SG * SE * 2);
    _Float16* enc_Whh16 = (_Float16*)carve((size_t)SG * SH * 2);
    _Float16* dec_Wih16 = (_Float16*)carve((size_t)SG * (SE + SH) * 2);
    _Float16* dec_Whh16 = (_Float16*)carve((size_t)SG * SH * 2);
    _Float16* lin_W16   = (_Float16*)carve((size_t)SV * SH * 2);
    _Float16* A_enc     = (_Float16*)carve((size_t)SB * ST * SE * 2);
    _Float16* A_dec     = (_Float16*)carve((size_t)SB * ST * (SE + SH) * 2);
    float*    gx_enc    = (float*)   carve((size_t)SB * ST * SG * 4);
    float*    gx_dec    = (float*)   carve((size_t)SB * ST * SG * 4);
    float*    h32[2]; _Float16* h16[2];
    h32[0] = (float*)carve((size_t)SB * SH * 4);
    h32[1] = (float*)carve((size_t)SB * SH * 4);
    h16[0] = (_Float16*)carve((size_t)SB * SH * 2);
    h16[1] = (_Float16*)carve((size_t)SB * SH * 2);
    _Float16* states16 = (_Float16*)carve((size_t)SB * ST * SH * 2);

    cvt_f32_f16<<<2048, 256, 0, stream>>>(enc_Wih, enc_Wih16, (size_t)SG * SE);
    cvt_f32_f16<<<2048, 256, 0, stream>>>(enc_Whh, enc_Whh16, (size_t)SG * SH);
    cvt_f32_f16<<<2048, 256, 0, stream>>>(dec_Wih, dec_Wih16, (size_t)SG * (SE + SH));
    cvt_f32_f16<<<2048, 256, 0, stream>>>(dec_Whh, dec_Whh16, (size_t)SG * SH);
    cvt_f32_f16<<<4096, 256, 0, stream>>>(lin_W,   lin_W16,   (size_t)SV * SH);

    const int M = SB * ST;  // 4096

    build_a_enc<<<M, 128, 0, stream>>>(x, enc_emb, A_enc);
    gemm_tn_wmma<<<dim3(SG / 128, M / 64), 256, 0, stream>>>(
        A_enc, enc_Wih16, enc_bih, gx_enc, M, SG, SE);

    init_h<<<(SB * SH + 255) / 256, 256, 0, stream>>>(h32[0], h16[0], nullptr);
    for (int t = 0; t < ST; ++t)
        gru_step_wmma<<<SH / 64, 256, 0, stream>>>(
            h16[t & 1], h32[t & 1], gx_enc, t, enc_Whh16, enc_bhh,
            h16[(t + 1) & 1], h32[(t + 1) & 1], nullptr);

    build_a_dec<<<M, 256, 0, stream>>>(labels, dec_emb, h16[0], A_dec);
    gemm_tn_wmma<<<dim3(SG / 128, M / 64), 256, 0, stream>>>(
        A_dec, dec_Wih16, dec_bih, gx_dec, M, SG, SE + SH);

    init_h<<<(SB * SH + 255) / 256, 256, 0, stream>>>(h32[0], h16[0], dec_init);
    for (int t = 0; t < ST; ++t)
        gru_step_wmma<<<SH / 64, 256, 0, stream>>>(
            h16[t & 1], h32[t & 1], gx_dec, t, dec_Whh16, dec_bhh,
            h16[(t + 1) & 1], h32[(t + 1) & 1], states16);

    gemm_tn_wmma<<<dim3(SV / 128, M / 64), 256, 0, stream>>>(
        states16, lin_W16, lin_b, out, M, SV, SH);
}
